// TemporalCrossAttention_51410758533168
// MI455X (gfx1250) — matrix operand order repaired and checked by reading, not compiled, not run on hardware
//
#include <hip/hip_runtime.h>

typedef __attribute__((ext_vector_type(16))) __bf16 v16bf;
typedef __attribute__((ext_vector_type(8)))  __bf16 v8bf;
typedef __attribute__((ext_vector_type(8)))  float  v8f;

#define TT     16
#define QD     320
#define INNER  512
#define NB     2
#define PX     328   // x LDS pitch (bf16 units), 656B = 41*16B, conflict-free
#define PQ     520   // qkv/O LDS pitch, 1040B = 65*16B, conflict-free
#define MT     (512*320)

// A-fragment (16x32 bf16): lane m=L&15 holds row m; kh=L>>4 selects K-half.
// elements 0..15 = K kk+kh*16..+15 (contiguous)
__device__ __forceinline__ v16bf load_a_frag(const __bf16* row, int kk, int kh) {
  v8bf lo = *reinterpret_cast<const v8bf*>(row + kk + kh * 16);
  v8bf hi = *reinterpret_cast<const v8bf*>(row + kk + kh * 16 + 8);
  v16bf r;
#pragma unroll
  for (int i = 0; i < 8; ++i) { r[i] = lo[i]; r[i + 8] = hi[i]; }
  return r;
}

// B-fragment (32x16 bf16): lane n=L&15 holds column n; lanes 0-15 K=0..15,
// lanes 16-31 K=16..31 (contiguous). p points at WT_row + kk + kh*16.
__device__ __forceinline__ v16bf load_b_frag(const __bf16* p) {
  v8bf lo = *reinterpret_cast<const v8bf*>(p);
  v8bf hi = *reinterpret_cast<const v8bf*>(p + 8);
  v16bf r;
#pragma unroll
  for (int i = 0; i < 8; ++i) { r[i] = lo[i]; r[i + 8] = hi[i]; }
  return r;
}

// ---- prep: f32 weights -> bf16, transposed for B-fragment loads -------------
// ws layout (bf16): [0..3*MT) = WqT,WkT,WvT each [512][320] (row n = W[:,n])
//                   [3*MT..4*MT) = WoT [320][512]
__global__ __launch_bounds__(256)
void tca_prep_weights(const float* __restrict__ Wq, const float* __restrict__ Wk,
                      const float* __restrict__ Wv, const float* __restrict__ Wo,
                      __bf16* __restrict__ ws) {
  int e = blockIdx.x * 256 + threadIdx.x;
  if (e < 3 * MT) {
    int mm = e / MT;
    int r  = e - mm * MT;
    int n  = r / QD;
    int kk = r - n * QD;
    const float* W = (mm == 0) ? Wq : ((mm == 1) ? Wk : Wv);
    ws[e] = (__bf16)W[kk * INNER + n];
  } else if (e < 4 * MT) {
    int r  = e - 3 * MT;
    int n  = r / INNER;
    int kk = r - n * INNER;
    ws[e] = (__bf16)Wo[kk * QD + n];
  }
}

// ---- fused QKV -> (2-entry) attention -> output projection ------------------
__global__ __launch_bounds__(256, 1)
void tca_fused_kernel(const float* __restrict__ x,
                      const float* __restrict__ bo,
                      const float* __restrict__ relk,
                      const float* __restrict__ relv,
                      const __bf16* __restrict__ wsw,
                      float* __restrict__ y) {
  __shared__ alignas(16) __bf16 s_qkv[3 * NB * TT * PQ];  // q | k | v, row-major per batch
  __shared__ alignas(16) __bf16 s_xo[NB * TT * PQ];       // x (pitch PX), reused as O (pitch PQ)
  __shared__ float s_rel[4 * 64];                         // relk15 | relk16 | relv15 | relv16

  const int tid  = threadIdx.x;
  const int bg   = blockIdx.x * NB;
  const int wave = tid >> 5;
  const int lane = tid & 31;
  const int m    = lane & 15;   // A row / B col / D col within tile
  const int kh   = lane >> 4;   // K-half selector

  // ---- phase 0: stage x (bf16) and the 4 needed rel rows ----
  for (int i = tid; i < NB * TT * QD; i += 256) {
    int b = i / (TT * QD);
    int rem = i - b * (TT * QD);
    int t = rem / QD, c = rem - t * QD;
    s_xo[(b * TT + t) * PX + c] = (__bf16)x[((size_t)(bg + b) * TT + t) * QD + c];
  }
  {
    int r = tid >> 6, d = tid & 63;                 // r: 0=relk15 1=relk16 2=relv15 3=relv16
    const float* tbl = (r < 2) ? relk : relv;
    int row = (r & 1) ? 16 : 15;
    s_rel[r * 64 + d] = tbl[row * 64 + d];
  }
  __syncthreads();

  // ---- phase 1: q,k,v = x @ W{q,k,v}. 96 tiles (3 mats x 32 n-tiles) / 8 waves ----
  for (int tt = 0; tt < 12; ++tt) {
    int tau = wave * 12 + tt;
    int mat = tau >> 5;                 // 0=q 1=k 2=v
    int n0  = (tau & 31) << 4;
    const __bf16* wrow = wsw + mat * MT + (n0 + m) * QD;
    v8f acc0 = {};
    v8f acc1 = {};
    for (int kk = 0; kk < QD; kk += 32) {
      v16bf B  = load_b_frag(wrow + kk + kh * 16);
      v16bf A0 = load_a_frag(&s_xo[(0 * TT + m) * PX], kk, kh);
      v16bf A1 = load_a_frag(&s_xo[(1 * TT + m) * PX], kk, kh);
      acc0 = __builtin_amdgcn_wmma_f32_16x16x32_bf16(false, A0, false, B, (short)0, acc0, false, false);
      acc1 = __builtin_amdgcn_wmma_f32_16x16x32_bf16(false, A1, false, B, (short)0, acc1, false, false);
    }
    __bf16* dst = s_qkv + mat * (NB * TT * PQ);
    int col = n0 + m;
#pragma unroll
    for (int r = 0; r < 8; ++r) {
      int row = r + 8 * kh;
      dst[(0 * TT + row) * PQ + col] = (__bf16)acc0[r];
      dst[(1 * TT + row) * PQ + col] = (__bf16)acc1[r];
    }
  }
  __syncthreads();

  // ---- phase 2: mask collapses to {s==t, s==t-1}; 2-entry softmax per (b,h,t) ----
  {
    int b = tid >> 7, h = (tid >> 4) & 7, t = tid & 15;
    const __bf16* qr = s_qkv + 0 * (NB * TT * PQ) + (b * TT + t) * PQ + h * 64;
    const __bf16* kb = s_qkv + 1 * (NB * TT * PQ) + b * TT * PQ;
    const __bf16* vb = s_qkv + 2 * (NB * TT * PQ) + b * TT * PQ;
    int tp = (t > 0) ? (t - 1) : 0;
    const __bf16* k1 = kb + t * PQ + h * 64;
    const __bf16* k0 = kb + tp * PQ + h * 64;
    float s1 = 0.f, s0 = 0.f;
#pragma unroll 8
    for (int d = 0; d < 64; ++d) {
      float qd = (float)qr[d];
      s1 += qd * ((float)k1[d] + s_rel[64 + d]);   // sim(t,t)  incl. rel_k[16]
      s0 += qd * ((float)k0[d] + s_rel[d]);        // sim(t,t-1) incl. rel_k[15]
    }
    s1 *= 0.125f;
    s0 *= 0.125f;
    float w0, w1;
    if (t == 0) { w0 = 0.f; w1 = 1.f; }
    else {
      float mx = fmaxf(s0, s1);
      float e0 = __expf(s0 - mx), e1 = __expf(s1 - mx);
      float z = e0 + e1;
      w0 = e0 / z;
      w1 = e1 / z;
    }
    const __bf16* v1 = vb + t * PQ + h * 64;
    const __bf16* v0 = vb + tp * PQ + h * 64;
    __bf16* orow = s_xo + (b * TT + t) * PQ + h * 64;   // O overwrites x region
#pragma unroll 8
    for (int d = 0; d < 64; ++d) {
      float o = w1 * ((float)v1[d] + s_rel[192 + d])    // + rel_v[16]
              + w0 * ((float)v0[d] + s_rel[128 + d]);   // + rel_v[15]
      orow[d] = (__bf16)o;
    }
  }
  __syncthreads();

  // ---- phase 3: y = O @ Wo + bo. 40 tiles (2 batches x 20 n-tiles) / 8 waves ----
  const __bf16* WoT = wsw + 3 * MT;
  for (int tt = 0; tt < 5; ++tt) {
    int tau = wave * 5 + tt;
    int b  = tau / 20;
    int n0 = (tau - b * 20) << 4;
    const __bf16* wrow = WoT + (n0 + m) * INNER;
    v8f acc = {};
    for (int kk = 0; kk < INNER; kk += 32) {
      v16bf B = load_b_frag(wrow + kk + kh * 16);
      v16bf A = load_a_frag(&s_xo[(b * TT + m) * PQ], kk, kh);
      acc = __builtin_amdgcn_wmma_f32_16x16x32_bf16(false, A, false, B, (short)0, acc, false, false);
    }
    float bias = bo[n0 + m];
    float* yb = y + ((size_t)(bg + b) * TT) * QD;
#pragma unroll
    for (int r = 0; r < 8; ++r) {
      int row = r + 8 * kh;
      yb[row * QD + n0 + m] = acc[r] + bias;
    }
  }
}

extern "C" void kernel_launch(void* const* d_in, const int* in_sizes, int n_in,
                              void* d_out, int out_size, void* d_ws, size_t ws_size,
                              hipStream_t stream) {
  const float* x    = (const float*)d_in[0];
  const float* Wq   = (const float*)d_in[1];
  const float* Wk   = (const float*)d_in[2];
  const float* Wv   = (const float*)d_in[3];
  const float* Wo   = (const float*)d_in[4];
  const float* bo   = (const float*)d_in[5];
  const float* relk = (const float*)d_in[6];
  const float* relv = (const float*)d_in[7];
  __bf16* wsw = (__bf16*)d_ws;
  float* y = (float*)d_out;

  int B = in_sizes[0] / (TT * QD);     // 8192
  int prep_blocks = (4 * MT + 255) / 256;
  tca_prep_weights<<<prep_blocks, 256, 0, stream>>>(Wq, Wk, Wv, Wo, wsw);
  tca_fused_kernel<<<B / NB, 256, 0, stream>>>(x, bo, relk, relv, wsw, y);
}
